// Transformer_59442347376855
// MI455X (gfx1250) — matrix operand order repaired and checked by reading, not compiled, not run on hardware
//
#include <hip/hip_runtime.h>
#include <hip/hip_bf16.h>
#include <math.h>

// ---------------------------------------------------------------------------
// MI455X (gfx1250) differential-attention transformer, bf16 WMMA everywhere.
// Compute-bound (~412 GFLOP, <1GB traffic) -> v_wmma_f32_16x16x32_bf16 path.
// GEMM: 128x64 block tile, 8 waves each owning a 32x32 (2x2 WMMA) C tile,
// operands staged in LDS via GLOBAL_LOAD_ASYNC_TO_LDS_B128 (ASYNCcnt).
// ---------------------------------------------------------------------------

typedef __bf16 bf16;
typedef __attribute__((ext_vector_type(16))) __bf16 v16bf;
typedef __attribute__((ext_vector_type(8)))  float   v8f;

static __device__ __forceinline__ v8f v8f_zero() {
    v8f z = {0.f, 0.f, 0.f, 0.f, 0.f, 0.f, 0.f, 0.f};
    return z;
}

#define WMMA_BF16(a, b, c) \
    __builtin_amdgcn_wmma_f32_16x16x32_bf16(false, (a), false, (b), (short)0, (c), false, false)

// ------------------------------ helpers ------------------------------------

__global__ __launch_bounds__(256) void cvt_f32_bf16_kernel(
    const float* __restrict__ src, bf16* __restrict__ dst, int n) {
    int i = blockIdx.x * blockDim.x + threadIdx.x;
    if (i < n) dst[i] = (bf16)src[i];
}

__global__ __launch_bounds__(256) void init_x_kernel(
    const float* __restrict__ x, float* __restrict__ xc, bf16* __restrict__ xbf, int n) {
    int i = blockIdx.x * blockDim.x + threadIdx.x;
    if (i < n) {
        float v = x[i];
        xc[i]  = v;
        xbf[i] = (bf16)v;
    }
}

// ------------------------------ GEMM ---------------------------------------
// C[m,n] = act( alpha*(sum_k A[m,k]*Bw[n,k] + bias[n]) ) + resid[m,n]
// A:[Mr,K] bf16 row-major, Bw:[N,K] bf16 row-major (nn.Linear layout).
// Block tile 128x64, BK=64 staged in LDS by async direct-to-LDS loads.
// Wave tile 32x32: 2 A-frags x 2 B-frags -> 4 WMMAs per K=32 step.

__global__ __launch_bounds__(256) void gemm_wmma_bf16_kernel(
    const bf16* __restrict__ A, const bf16* __restrict__ Bw,
    const float* __restrict__ bias, const float* __restrict__ dw_ptr, int alpha_mode,
    const float* __restrict__ resid, float* __restrict__ Cf, bf16* __restrict__ Cb,
    int Mr, int N, int K, int do_gelu) {
    __shared__ bf16 As[128][64];  // 16 KB
    __shared__ bf16 Bs[64][64];   //  8 KB

    const int t    = threadIdx.x;
    const int wid  = t >> 5;
    const int lane = t & 31;
    const int lrow = lane & 15;
    const int lhi  = lane >> 4;
    const int wm   = (wid & 3) * 32;   // wave tile row inside 128
    const int wn   = (wid >> 2) * 32;  // wave tile col inside 64
    const int m0   = blockIdx.y * 128;
    const int n0   = blockIdx.x * 64;

    // Async loaders: A 16KB/256thr = 64B each; B 8KB/256thr = 32B each.
    const int arow = t >> 1, achk = (t & 1) * 32;  // 32 bf16 = 64B
    const int brow = t >> 2, bchk = (t & 3) * 16;  // 16 bf16 = 32B
    const unsigned aLds = (unsigned)(size_t)&As[arow][achk];
    const unsigned bLds = (unsigned)(size_t)&Bs[brow][bchk];
    const bf16* agp = A + (size_t)(m0 + arow) * K + achk;
    const bf16* bgp = Bw + (size_t)(n0 + brow) * K + bchk;

    v8f acc[2][2];
    acc[0][0] = v8f_zero(); acc[0][1] = v8f_zero();
    acc[1][0] = v8f_zero(); acc[1][1] = v8f_zero();

    for (int k0 = 0; k0 < K; k0 += 64) {
        {
            unsigned long long ga = (unsigned long long)(size_t)(agp + k0);
            unsigned long long gb = (unsigned long long)(size_t)(bgp + k0);
            // Direct global->LDS async copies (ASYNCcnt). INST_OFFSET is added
            // to both the LDS and global addresses (ISA 15.18.3), so one base
            // VGPR per side covers the whole per-thread chunk.
            asm volatile(
                "global_load_async_to_lds_b128 %0, %2, off\n\t"
                "global_load_async_to_lds_b128 %0, %2, off offset:16\n\t"
                "global_load_async_to_lds_b128 %0, %2, off offset:32\n\t"
                "global_load_async_to_lds_b128 %0, %2, off offset:48\n\t"
                "global_load_async_to_lds_b128 %1, %3, off\n\t"
                "global_load_async_to_lds_b128 %1, %3, off offset:16"
                :: "v"(aLds), "v"(bLds), "v"(ga), "v"(gb)
                : "memory");
            if (k0 + 64 < K) {  // warm L2 for the next stage
                __builtin_prefetch(agp + k0 + 64, 0, 0);
                __builtin_prefetch(bgp + k0 + 64, 0, 0);
            }
        }
        asm volatile("s_wait_asynccnt 0x0" ::: "memory");
        __syncthreads();

        #pragma unroll
        for (int kk = 0; kk < 64; kk += 32) {
            v16bf a[2], b[2];
            const int kbA = kk + 8 * lhi;
            const int kbB = kk + 8 * lhi;
            #pragma unroll
            for (int i = 0; i < 2; ++i)
                #pragma unroll
                for (int e = 0; e < 16; ++e)
                    a[i][e] = As[wm + i * 16 + lrow][kbA + (e < 8 ? e : e + 8)];
            #pragma unroll
            for (int j = 0; j < 2; ++j)
                #pragma unroll
                for (int e = 0; e < 16; ++e)
                    b[j][e] = Bs[wn + j * 16 + lrow][kbB + (e < 8 ? e : e + 8)];
            acc[0][0] = WMMA_BF16(a[0], b[0], acc[0][0]);
            acc[0][1] = WMMA_BF16(a[0], b[1], acc[0][1]);
            acc[1][0] = WMMA_BF16(a[1], b[0], acc[1][0]);
            acc[1][1] = WMMA_BF16(a[1], b[1], acc[1][1]);
        }
        __syncthreads();
    }

    // Epilogue. C layout: VGPR r -> row r + 8*lhi, col = lane&15.
    float alpha = 1.f;
    if (alpha_mode == 1) alpha = 1.f - dw_ptr[0];
    #pragma unroll
    for (int j = 0; j < 2; ++j) {
        const int   gcol = n0 + wn + j * 16 + lrow;
        const float bval = bias ? bias[gcol] : 0.f;
        #pragma unroll
        for (int i = 0; i < 2; ++i) {
            #pragma unroll
            for (int r = 0; r < 8; ++r) {
                const int grow = m0 + wm + i * 16 + r + 8 * lhi;
                float v = (acc[i][j][r] + bval) * alpha;
                if (do_gelu) v = 0.5f * v * (1.f + erff(v * 0.70710678118654752f));
                const size_t idx = (size_t)grow * N + gcol;
                if (resid) v += resid[idx];
                if (Cf) Cf[idx] = v;
                if (Cb) Cb[idx] = (bf16)v;
            }
        }
    }
}

// ------------------------ differential flash attention ---------------------
// One wave per (b, h, 16-query tile). F=32 per component == WMMA K depth.
// Two running softmaxes (components 0/1), probabilities staged in LDS as
// bf16 in A-fragment order, PV via WMMA. Combine o0/l0 - dw*o1/l1 at the end.

__global__ __launch_bounds__(32) void diff_flash_attn_kernel(
    const bf16* __restrict__ q, const bf16* __restrict__ k, const bf16* __restrict__ v,
    const float* __restrict__ dw_ptr, float* __restrict__ o, int S, int D, int H) {
    const int qt = blockIdx.x, h = blockIdx.y, b = blockIdx.z;
    const int lane = threadIdx.x;
    const int lrow = lane & 15, lhi = lane >> 4;
    const int F = 32, E = 64;
    const size_t tokBase = (size_t)b * S;

    __shared__ bf16 P[2][16][32];

    // Q fragments (A layout), one per score component.
    v16bf qa[2];
    {
        const int tok = qt * 16 + lrow;
        const bf16* qp = q + (tokBase + tok) * D + h * E;
        const int kb = 8 * lhi;
        #pragma unroll
        for (int comp = 0; comp < 2; ++comp)
            #pragma unroll
            for (int e = 0; e < 16; ++e)
                qa[comp][e] = qp[comp * F + kb + (e < 8 ? e : e + 8)];
    }

    float mmax[2][8], lsum[2][8];
    v8f   acc[2][4];
    #pragma unroll
    for (int comp = 0; comp < 2; ++comp) {
        #pragma unroll
        for (int r = 0; r < 8; ++r) { mmax[comp][r] = -1e30f; lsum[comp][r] = 0.f; }
        #pragma unroll
        for (int ct = 0; ct < 4; ++ct) acc[comp][ct] = v8f_zero();
    }
    const float scale = 0.17677669529663689f;  // 32^-0.5

    for (int jt = 0; jt < S; jt += 32) {
        // Scores: two 16x16 tiles per component.
        v8f sc[2][2];
        #pragma unroll
        for (int sub = 0; sub < 2; ++sub) {
            const int kt = jt + sub * 16;
            const bf16* kp = k + (tokBase + kt + lrow) * D + h * E + 8 * lhi;
            #pragma unroll
            for (int comp = 0; comp < 2; ++comp) {
                v16bf kf;
                #pragma unroll
                for (int e = 0; e < 16; ++e) kf[e] = kp[comp * F + (e < 8 ? e : e + 8)];
                sc[comp][sub] = WMMA_BF16(qa[comp], kf, v8f_zero());
            }
        }
        // Online softmax update for both components over the 32 new keys.
        #pragma unroll
        for (int comp = 0; comp < 2; ++comp) {
            #pragma unroll
            for (int r = 0; r < 8; ++r) {
                float a0 = sc[comp][0][r] * scale;
                float a1 = sc[comp][1][r] * scale;
                float mx = fmaxf(a0, a1);
                #pragma unroll
                for (int off = 1; off < 16; off <<= 1)
                    mx = fmaxf(mx, __shfl_xor(mx, off, 32));
                const float nm   = fmaxf(mmax[comp][r], mx);
                const float corr = __expf(mmax[comp][r] - nm);
                mmax[comp][r] = nm;
                const float p0 = __expf(a0 - nm);
                const float p1 = __expf(a1 - nm);
                float rs = p0 + p1;
                #pragma unroll
                for (int off = 1; off < 16; off <<= 1)
                    rs += __shfl_xor(rs, off, 32);
                lsum[comp][r] = lsum[comp][r] * corr + rs;
                #pragma unroll
                for (int ct = 0; ct < 4; ++ct) acc[comp][ct][r] *= corr;
                P[comp][r + 8 * lhi][lrow]      = (bf16)p0;
                P[comp][r + 8 * lhi][16 + lrow] = (bf16)p1;
            }
        }
        __syncthreads();
        // PV: P(16x32) x V(32x64) via 4 column tiles, shared V fragment.
        v16bf pa[2];
        {
            const int kb = 8 * lhi;
            #pragma unroll
            for (int comp = 0; comp < 2; ++comp)
                #pragma unroll
                for (int e = 0; e < 16; ++e)
                    pa[comp][e] = P[comp][lrow][kb + (e < 8 ? e : e + 8)];
        }
        #pragma unroll
        for (int ct = 0; ct < 4; ++ct) {
            v16bf vf;
            const bf16* vp = v + (tokBase + jt + 8 * lhi) * D + h * E + ct * 16 + lrow;
            #pragma unroll
            for (int e = 0; e < 16; ++e) vf[e] = vp[(size_t)(e < 8 ? e : e + 8) * D];
            acc[0][ct] = WMMA_BF16(pa[0], vf, acc[0][ct]);
            acc[1][ct] = WMMA_BF16(pa[1], vf, acc[1][ct]);
        }
        __syncthreads();
    }

    const float dw = dw_ptr[0];
    #pragma unroll
    for (int r = 0; r < 8; ++r) {
        const int tok = qt * 16 + r + 8 * lhi;
        const float i0 = 1.f / lsum[0][r];
        const float i1 = 1.f / lsum[1][r];
        float* op = o + (tokBase + tok) * D + h * E;
        #pragma unroll
        for (int ct = 0; ct < 4; ++ct)
            op[ct * 16 + lrow] = acc[0][ct][r] * i0 - dw * acc[1][ct][r] * i1;
    }
}

// ------------------------------ normalizations -----------------------------

__global__ __launch_bounds__(256) void rms_norm_kernel(
    const float* __restrict__ o, const float* __restrict__ rw,
    bf16* __restrict__ outb, int D) {
    __shared__ float red[256];
    const int tok = blockIdx.x, t = threadIdx.x;
    const float* p = o + (size_t)tok * D;
    float s = 0.f;
    for (int i = t; i < D; i += 256) { float v = p[i]; s += v * v; }
    red[t] = s;
    __syncthreads();
    for (int st = 128; st > 0; st >>= 1) {
        if (t < st) red[t] += red[t + st];
        __syncthreads();
    }
    const float inv = rsqrtf(red[0] / D + 1.1920929e-07f);
    for (int i = t; i < D; i += 256)
        outb[(size_t)tok * D + i] = (bf16)(p[i] * inv * rw[i]);
}

__global__ __launch_bounds__(256) void layernorm_kernel(
    const float* __restrict__ x, const float* __restrict__ g, const float* __restrict__ b,
    float* __restrict__ outf, bf16* __restrict__ outb, int D) {
    __shared__ float rs[256], rq[256];
    const int tok = blockIdx.x, t = threadIdx.x;
    const float* p = x + (size_t)tok * D;
    float s = 0.f, q = 0.f;
    for (int i = t; i < D; i += 256) { float v = p[i]; s += v; q += v * v; }
    rs[t] = s; rq[t] = q;
    __syncthreads();
    for (int st = 128; st > 0; st >>= 1) {
        if (t < st) { rs[t] += rs[t + st]; rq[t] += rq[t + st]; }
        __syncthreads();
    }
    const float mean = rs[0] / D;
    const float var  = rq[0] / D - mean * mean;
    const float inv  = rsqrtf(var + 1e-5f);
    for (int i = t; i < D; i += 256) {
        float y = (p[i] - mean) * inv * g[i] + b[i];
        if (outf) outf[(size_t)tok * D + i] = y;
        if (outb) outb[(size_t)tok * D + i] = (bf16)y;
    }
}

// ------------------------------ host orchestration -------------------------

extern "C" void kernel_launch(void* const* d_in, const int* in_sizes, int n_in,
                              void* d_out, int out_size, void* d_ws, size_t ws_size,
                              hipStream_t stream) {
    (void)in_sizes; (void)n_in; (void)out_size; (void)ws_size;
    constexpr int L = 4, D = 512, H = 8, M = 2048, B = 4, S = 2048;
    constexpr int Mr = B * S;  // 8192 tokens

    const float* x     = (const float*)d_in[0];
    const float* Wq    = (const float*)d_in[1];
    const float* Wk    = (const float*)d_in[2];
    const float* Wv    = (const float*)d_in[3];
    const float* rms_w = (const float*)d_in[4];
    const float* Wo    = (const float*)d_in[5];
    const float* bo    = (const float*)d_in[6];
    const float* diffw = (const float*)d_in[7];
    const float* g_ffn = (const float*)d_in[8];
    const float* b_ffn = (const float*)d_in[9];
    const float* W1    = (const float*)d_in[10];
    const float* b1    = (const float*)d_in[11];
    const float* W2    = (const float*)d_in[12];
    const float* b2    = (const float*)d_in[13];
    const float* g_f   = (const float*)d_in[14];
    const float* b_f   = (const float*)d_in[15];

    // Workspace carve-up (~136 MiB total).
    char* wp = (char*)d_ws;
    auto carve = [&](size_t bytes) {
        void* p = (void*)wp;
        wp += (bytes + 255) & ~(size_t)255;
        return p;
    };
    float* xc   = (float*)carve((size_t)Mr * D * 4);
    bf16*  xbf  = (bf16*) carve((size_t)Mr * D * 2);
    bf16*  qbf  = (bf16*) carve((size_t)Mr * D * 2);
    bf16*  kbf  = (bf16*) carve((size_t)Mr * D * 2);
    bf16*  vbf  = (bf16*) carve((size_t)Mr * D * 2);
    float* obuf = (float*)carve((size_t)Mr * D * 4);
    bf16*  obf  = (bf16*) carve((size_t)Mr * D * 2);
    bf16*  lnbf = (bf16*) carve((size_t)Mr * D * 2);
    bf16*  hbf  = (bf16*) carve((size_t)Mr * M * 2);
    bf16*  Wq_b = (bf16*) carve((size_t)L * D * D * 2);
    bf16*  Wk_b = (bf16*) carve((size_t)L * D * D * 2);
    bf16*  Wv_b = (bf16*) carve((size_t)L * D * D * 2);
    bf16*  Wo_b = (bf16*) carve((size_t)L * D * D * 2);
    bf16*  W1_b = (bf16*) carve((size_t)L * M * D * 2);
    bf16*  W2_b = (bf16*) carve((size_t)L * D * M * 2);

    auto cvt = [&](const float* src, bf16* dst, int n) {
        cvt_f32_bf16_kernel<<<(n + 255) / 256, 256, 0, stream>>>(src, dst, n);
    };
    cvt(Wq, Wq_b, L * D * D);
    cvt(Wk, Wk_b, L * D * D);
    cvt(Wv, Wv_b, L * D * D);
    cvt(Wo, Wo_b, L * D * D);
    cvt(W1, W1_b, L * M * D);
    cvt(W2, W2_b, L * D * M);
    init_x_kernel<<<(Mr * D + 255) / 256, 256, 0, stream>>>(x, xc, xbf, Mr * D);

    auto gemm = [&](const bf16* A, const bf16* Bw, const float* bias,
                    const float* dwp, int amode, const float* resid,
                    float* Cf, bf16* Cb, int N, int K, int gelu) {
        dim3 grid(N / 64, Mr / 128);
        gemm_wmma_bf16_kernel<<<grid, 256, 0, stream>>>(
            A, Bw, bias, dwp, amode, resid, Cf, Cb, Mr, N, K, gelu);
    };

    for (int l = 0; l < L; ++l) {
        // QKV projections (bf16 outputs feed attention fragments directly).
        gemm(xbf, Wq_b + (size_t)l * D * D, nullptr, nullptr, 0, nullptr, nullptr, qbf, D, D, 0);
        gemm(xbf, Wk_b + (size_t)l * D * D, nullptr, nullptr, 0, nullptr, nullptr, kbf, D, D, 0);
        gemm(xbf, Wv_b + (size_t)l * D * D, nullptr, nullptr, 0, nullptr, nullptr, vbf, D, D, 0);
        // Differential flash attention -> o (fp32).
        dim3 agrid(S / 16, H, B);
        diff_flash_attn_kernel<<<agrid, 32, 0, stream>>>(qbf, kbf, vbf, diffw + l, obuf, S, D, H);
        // Fused per-token RMSNorm over (H, D/H) with weight rms_w[l].
        rms_norm_kernel<<<Mr, 256, 0, stream>>>(obuf, rms_w + (size_t)l * D, obf, D);
        // Wo projection: xc = (obf @ Wo^T + bo) * (1 - dw) + xc.
        gemm(obf, Wo_b + (size_t)l * D * D, bo + (size_t)l * D, diffw + l, 1, xc, xc, nullptr, D, D, 0);
        // FFN: h = gelu(LN(xc) @ W1^T + b1); xc = h @ W2^T + b2 + xc.
        layernorm_kernel<<<Mr, 256, 0, stream>>>(xc, g_ffn + (size_t)l * D, b_ffn + (size_t)l * D,
                                                 nullptr, lnbf, D);
        gemm(lnbf, W1_b + (size_t)l * M * D, b1 + (size_t)l * M, nullptr, 0, nullptr,
             nullptr, hbf, M, D, 1);
        gemm(hbf, W2_b + (size_t)l * D * M, b2 + (size_t)l * D, nullptr, 0, xc,
             xc, xbf, D, M, 0);
    }
    // Final layernorm -> fp32 output.
    layernorm_kernel<<<Mr, 256, 0, stream>>>(xc, g_f, b_f, (float*)d_out, nullptr, D);
}
